// ODEBlock_88381837017192
// MI455X (gfx1250) — compile-verified, matrix operand order checked by reading, not run on hardware
//
#include <hip/hip_runtime.h>
#include <hip/hip_bf16.h>
#include <stddef.h>

// ---------------------------------------------------------------------------
// Neural-ODE block for MI455X (gfx1250): fixed-step RK4 over
//   f(y,t) = conv3x3_SAME(concat(t, sigmoid(y)), W) + b
// Conv = implicit GEMM on bf16 WMMA (v_wmma_f32_16x16x32_bf16).
// Weights are staged into LDS with gfx1250 async global->LDS DMA
// (global_load_async_to_lds_b128 / s_wait_asynccnt), double-buffered per tap.
// ---------------------------------------------------------------------------

typedef __attribute__((ext_vector_type(16))) __bf16 v16bf;
typedef __attribute__((ext_vector_type(8)))  __bf16 v8bf;
typedef __attribute__((ext_vector_type(8)))  float  v8f;

#define BATCH 32
#define HDIM  64
#define WDIM  64
#define CIN   128      // sigmoid channels (time channel handled analytically)
#define COUT  128
#define TW    66       // 64 pixels + 1 halo each side
#define CSTR  136      // padded LDS channel stride (bank-conflict friendly)
#define NSTEPS 8

static __device__ __forceinline__ __bf16 f2bf(float f) {
    union { float f; unsigned u; } a; a.f = f;
    unsigned r = a.u + 0x7FFFu + ((a.u >> 16) & 1u);   // round-to-nearest-even
    unsigned short h = (unsigned short)(r >> 16);
    return __builtin_bit_cast(__bf16, h);
}

static __device__ __forceinline__ float sigm(float x) {
    return 1.0f / (1.0f + __expf(-x));
}

// Issue one tap's 128x128 bf16 weight tile (dense [co][ci], 256B rows) into a
// padded LDS buffer (CSTR-stride rows) as 2048 x 16B async DMA chunks.
// 256 threads x 8 chunks; consecutive lanes move consecutive 16B chunks.
static __device__ __forceinline__ void issue_wtap_async(
    const __bf16* __restrict__ gsrc, __bf16* lbuf, int tid)
{
    #pragma unroll
    for (int j = 0; j < 8; ++j) {
        const int c   = j * 256 + tid;       // chunk id 0..2047
        const int row = c >> 4;              // co (0..127)
        const int cik = c & 15;              // 16B chunk within row
        const unsigned ldsoff =
            (unsigned)(size_t)(lbuf + row * CSTR + cik * 8);
        const unsigned long long ga =
            (unsigned long long)(size_t)(gsrc + row * CIN + cik * 8);
        asm volatile("global_load_async_to_lds_b128 %0, %1, off"
                     :: "v"(ldsoff), "v"(ga) : "memory");
    }
}

// ---------------------------------------------------------------------------
// Prep: transpose fp32 HWIO weights (3,3,129,128) -> bf16 Wt[tap][co][ci]
// (ci = original channel ci+1, i.e. skip the time channel).
// ---------------------------------------------------------------------------
__global__ void prep_weights(const float* __restrict__ w, __bf16* __restrict__ Wt) {
    int e = blockIdx.x * blockDim.x + threadIdx.x;
    if (e >= 9 * COUT * CIN) return;
    int tap = e / (COUT * CIN);
    int rem = e - tap * (COUT * CIN);
    int co  = rem / CIN;
    int ci  = rem - co * CIN;
    Wt[e] = f2bf(w[((size_t)tap * 129 + (ci + 1)) * COUT + co]);
}

__global__ void copy4(const float* __restrict__ src, float* __restrict__ dst) {
    size_t i = (size_t)blockIdx.x * blockDim.x + threadIdx.x;
    ((float4*)dst)[i] = ((const float4*)src)[i];
}

// ---------------------------------------------------------------------------
// Fused conv + RK4 stage update.
//   mode 0 (k1): acc = k;      sOut = y + h/2 * k
//   mode 1 (k2): acc += 2k;    sOut = y + h/2 * k
//   mode 2 (k3): acc += 2k;    sOut = y + h   * k
//   mode 3 (k4): y  += h/6 * (acc + k)
// One block per (b,h) row: M = 64 pixels, N = 128 channels, 8 wave32s.
// ---------------------------------------------------------------------------
__global__ void __launch_bounds__(256)
ode_conv_stage(const float* __restrict__ sin,   // state fed through sigmoid
               float* __restrict__ y,
               float* __restrict__ accg,
               float* __restrict__ sOut,
               const __bf16* __restrict__ Wt,
               const float* __restrict__ wfull, // original fp32 weights (time taps)
               const float* __restrict__ bias,
               float tstage, float hstep, int mode)
{
    __shared__ __bf16 tile[3 * TW * CSTR];       // sigmoid'd input halo tile
    __shared__ __bf16 wbuf[2][COUT * CSTR];      // double-buffered weight tap

    const int b   = blockIdx.x >> 6;
    const int h   = blockIdx.x & 63;
    const int tid = threadIdx.x;

    // Kick off async DMA of tap 0 weights; overlaps the input-tile fill.
    issue_wtap_async(Wt, &wbuf[0][0], tid);

    // -------- stage input -> sigmoid -> bf16 halo tile in LDS --------------
    if (tid < 3 * TW) {
        const int r  = tid / TW;
        const int c  = tid - r * TW;
        const int gh = h + r - 1;
        const int gw = c - 1;
        __bf16* dst = &tile[(r * TW + c) * CSTR];
        if (gh >= 0 && gh < HDIM && gw >= 0 && gw < WDIM) {
            const float* src = sin + (((size_t)b * HDIM + gh) * WDIM + gw) * CIN;
            #pragma unroll 8
            for (int ch = 0; ch < CIN; ch += 4) {
                float4 v = *(const float4*)(src + ch);
                dst[ch + 0] = f2bf(sigm(v.x));
                dst[ch + 1] = f2bf(sigm(v.y));
                dst[ch + 2] = f2bf(sigm(v.z));
                dst[ch + 3] = f2bf(sigm(v.w));
            }
        } else {
            const __bf16 z = f2bf(0.0f);   // SAME padding (post-concat zeros)
            #pragma unroll 8
            for (int ch = 0; ch < CIN; ++ch) dst[ch] = z;
        }
    }

    // -------- implicit GEMM: 9 taps x 4 k-chunks of 32 ---------------------
    const int wave  = tid >> 5;
    const int lane  = tid & 31;
    const int laneM = lane & 15;
    const int khalf = lane >> 4;           // 0: K 0-7/16-23, 1: K 8-15/24-31
    const int mtile = wave & 3;            // 4 M-tiles of 16 pixels
    const int nbase = (wave >> 2) * 64;    // 2 wave-groups cover N=128

    v8f acc[4] = {v8f{}, v8f{}, v8f{}, v8f{}};

    for (int tap = 0; tap < 9; ++tap) {
        // My async writes for this tap are done; barrier makes everyone's
        // writes visible and guarantees previous-tap reads are consumed
        // before the other buffer gets overwritten below.
        asm volatile("s_wait_asynccnt 0x0" ::: "memory");
        __syncthreads();
        if (tap + 1 < 9)
            issue_wtap_async(Wt + (size_t)(tap + 1) * COUT * CIN,
                             &wbuf[(tap + 1) & 1][0], tid);

        const int ky = tap / 3;
        const int kx = tap - ky * 3;
        const __bf16* arow = &tile[((ky * TW) + (mtile * 16 + laneM + kx)) * CSTR];
        const __bf16* wb   = &wbuf[tap & 1][0];
        #pragma unroll
        for (int chunk = 0; chunk < 4; ++chunk) {
            const int c0 = chunk * 32 + khalf * 8;
            v8bf alo = *(const v8bf*)(arow + c0);
            v8bf ahi = *(const v8bf*)(arow + c0 + 16);
            v16bf afrag = __builtin_shufflevector(alo, ahi,
                0, 1, 2, 3, 4, 5, 6, 7, 8, 9, 10, 11, 12, 13, 14, 15);
            #pragma unroll
            for (int nt = 0; nt < 4; ++nt) {
                const __bf16* bp = wb + (nbase + nt * 16 + laneM) * CSTR + c0;
                v8bf blo = *(const v8bf*)(bp);
                v8bf bhi = *(const v8bf*)(bp + 16);
                v16bf bfrag = __builtin_shufflevector(blo, bhi,
                    0, 1, 2, 3, 4, 5, 6, 7, 8, 9, 10, 11, 12, 13, 14, 15);
                acc[nt] = __builtin_amdgcn_wmma_f32_16x16x32_bf16(
                    false, afrag, false, bfrag, (short)0, acc[nt], false, false);
            }
        }
    }

    // -------- epilogue: bias + time channel + RK stage algebra -------------
    const int pixrowbase = (b * HDIM + h) * WDIM;
    #pragma unroll
    for (int nt = 0; nt < 4; ++nt) {
        const int co = nbase + nt * 16 + laneM;
        const float bv = bias[co];
        // time-channel tap sums per kx over valid ky (h-border handling)
        float ts0 = 0.f, ts1 = 0.f, ts2 = 0.f;
        #pragma unroll
        for (int ky = 0; ky < 3; ++ky) {
            const int gh = h + ky - 1;
            if (gh >= 0 && gh < HDIM) {
                ts0 += wfull[((size_t)(ky * 3 + 0) * 129) * COUT + co];
                ts1 += wfull[((size_t)(ky * 3 + 1) * 129) * COUT + co];
                ts2 += wfull[((size_t)(ky * 3 + 2) * 129) * COUT + co];
            }
        }
        #pragma unroll
        for (int r = 0; r < 8; ++r) {
            const int m = mtile * 16 + (khalf ? (8 + r) : r);   // pixel (w coord)
            const float tsum = ts1 + (m > 0 ? ts0 : 0.f) + (m < WDIM - 1 ? ts2 : 0.f);
            const float kval = acc[nt][r] + bv + tstage * tsum;
            const size_t idx = ((size_t)(pixrowbase + m)) * COUT + co;
            if (mode == 0) {
                accg[idx] = kval;
                sOut[idx] = y[idx] + 0.5f * hstep * kval;
            } else if (mode == 1) {
                accg[idx] += 2.0f * kval;
                sOut[idx] = y[idx] + 0.5f * hstep * kval;
            } else if (mode == 2) {
                accg[idx] += 2.0f * kval;
                sOut[idx] = y[idx] + hstep * kval;
            } else {
                y[idx] += (hstep / 6.0f) * (accg[idx] + kval);
            }
        }
    }
}

// ---------------------------------------------------------------------------
extern "C" void kernel_launch(void* const* d_in, const int* in_sizes, int n_in,
                              void* d_out, int out_size, void* d_ws, size_t ws_size,
                              hipStream_t stream) {
    (void)in_sizes; (void)n_in; (void)out_size; (void)ws_size;
    const float* x    = (const float*)d_in[0];
    const float* w    = (const float*)d_in[1];
    const float* bias = (const float*)d_in[2];
    float* out = (float*)d_out;

    const size_t NELEM = (size_t)BATCH * HDIM * WDIM * CIN;   // 16,777,216
    char* ws = (char*)d_ws;
    float*  y    = (float*)(ws);
    float*  accg = (float*)(ws + 1 * NELEM * sizeof(float));
    float*  sA   = (float*)(ws + 2 * NELEM * sizeof(float));
    float*  sB   = (float*)(ws + 3 * NELEM * sizeof(float));
    __bf16* Wt   = (__bf16*)(ws + 4 * NELEM * sizeof(float));

    prep_weights<<<(9 * COUT * CIN + 255) / 256, 256, 0, stream>>>(w, Wt);
    copy4<<<(unsigned)(NELEM / 4 / 256), 256, 0, stream>>>(x, y);

    const int nconv = BATCH * HDIM;      // 2048 blocks per dynamics eval
    const float hstep = 1.0f / (float)NSTEPS;
    for (int s = 0; s < NSTEPS; ++s) {
        const float t0 = s * hstep;
        ode_conv_stage<<<nconv, 256, 0, stream>>>(y,  y, accg, sA, Wt, w, bias,
                                                  t0,               hstep, 0);
        ode_conv_stage<<<nconv, 256, 0, stream>>>(sA, y, accg, sB, Wt, w, bias,
                                                  t0 + 0.5f * hstep, hstep, 1);
        ode_conv_stage<<<nconv, 256, 0, stream>>>(sB, y, accg, sA, Wt, w, bias,
                                                  t0 + 0.5f * hstep, hstep, 2);
        ode_conv_stage<<<nconv, 256, 0, stream>>>(sA, y, accg, sB, Wt, w, bias,
                                                  t0 + hstep,        hstep, 3);
    }
    copy4<<<(unsigned)(NELEM / 4 / 256), 256, 0, stream>>>(y, out);
}
